// BatchPrototypeLoss_56435870269925
// MI455X (gfx1250) — matrix operand order repaired and checked by reading, not compiled, not run on hardware
//
#include <hip/hip_runtime.h>

#define N_ROWS 65536
#define DIM    256
#define NCLS   512

typedef float  v8f  __attribute__((ext_vector_type(8)));
typedef __bf16 v8bf __attribute__((ext_vector_type(8)));
typedef __bf16 v16bf __attribute__((ext_vector_type(16)));

// ---- bf16 helpers (storage-only; no reliance on __bf16 arithmetic) ----
__device__ __forceinline__ __bf16 u2bf(unsigned short v) {
    union { unsigned short u; __bf16 b; } cv; cv.u = v; return cv.b;
}
__device__ __forceinline__ float bf2f(__bf16 b) {
    union { __bf16 b; unsigned short u; } cv; cv.b = b;
    return __uint_as_float((unsigned)cv.u << 16);
}
__device__ __forceinline__ unsigned short bfbits_trunc(float x) {
    return (unsigned short)(__float_as_uint(x) >> 16);
}
__device__ __forceinline__ __bf16 f2bf_rne(float x) {
    unsigned u = __float_as_uint(x);
    u += 0x7FFFu + ((u >> 16) & 1u);
    return u2bf((unsigned short)(u >> 16));
}

// ---------------- K0: zero sums+counts region and the scalar output ----------------
__global__ void __launch_bounds__(256) zero_kernel(float* p, int n, float* out) {
    int i = blockIdx.x * 256 + threadIdx.x;
    if (i < n) p[i] = 0.0f;
    if (i == 0) out[0] = 0.0f;
}

// ---------------- K1: segment-sum via LDS accumulation ----------------
// grid = (16 col-chunks of 16, 16 row-chunks of 4096), block = 256
__global__ void __launch_bounds__(256) scatter_kernel(const float* __restrict__ inputs,
                                                      const int* __restrict__ targets,
                                                      float* __restrict__ sums) {
    __shared__ float acc[NCLS * 16];   // 32 KB
    const int t = threadIdx.x;
    for (int i = t; i < NCLS * 16; i += 256) acc[i] = 0.0f;
    __syncthreads();

    const int colBase = blockIdx.x * 16;
    const int rowBase = blockIdx.y * 4096;
    const int jc   = t & 15;
    const int rsub = t >> 4;
    const int col  = colBase + jc;
    for (int rr = 0; rr < 4096; rr += 16) {
        const int row = rowBase + rr + rsub;
        const int tg  = targets[row];
        const float v = inputs[(size_t)row * DIM + col];
        atomicAdd(&acc[tg * 16 + jc], v);           // ds_add_f32
    }
    __syncthreads();
    for (int i = t; i < NCLS * 16; i += 256) {
        const int c = i >> 4, j = i & 15;
        atomicAdd(&sums[(size_t)c * DIM + colBase + j], acc[i]);
    }
}

// ---------------- K1b: class counts via LDS histogram ----------------
__global__ void __launch_bounds__(256) count_kernel(const int* __restrict__ targets,
                                                    float* __restrict__ counts) {
    __shared__ float c[NCLS];
    const int t = threadIdx.x;
    c[t] = 0.0f; c[t + 256] = 0.0f;
    __syncthreads();
    const int base = blockIdx.x * 1024;
    for (int i = 0; i < 4; ++i)
        atomicAdd(&c[targets[base + i * 256 + t]], 1.0f);
    __syncthreads();
    atomicAdd(&counts[t], c[t]);
    atomicAdd(&counts[t + 256], c[t + 256]);
}

// ---------------- K2: prototypes -> split-bf16 hi/lo planes + per-class norms ----------------
// grid = 512 (one class per block), block = 256
__global__ void __launch_bounds__(256) proto_kernel(const float* __restrict__ sums,
                                                    const float* __restrict__ counts,
                                                    __bf16* __restrict__ phi,
                                                    __bf16* __restrict__ plo,
                                                    float* __restrict__ npn) {
    const int c = blockIdx.x, t = threadIdx.x;
    const float cnt = counts[c];
    const float v = sums[(size_t)c * DIM + t] / cnt;
    const unsigned short hb = bfbits_trunc(v);       // hi = truncate (prefix of mantissa)
    const float hf = __uint_as_float((unsigned)hb << 16);
    phi[(size_t)c * DIM + t] = u2bf(hb);
    plo[(size_t)c * DIM + t] = f2bf_rne(v - hf);     // residual, RNE
    float sq = v * v;
    for (int off = 16; off > 0; off >>= 1) sq += __shfl_xor(sq, off, 32);
    __shared__ float red[8];
    if ((t & 31) == 0) red[t >> 5] = sq;
    __syncthreads();
    if (t == 0) {
        float s = 0.0f;
        for (int i = 0; i < 8; ++i) s += red[i];
        npn[c] = sqrtf(s);
    }
}

// ---------------- K3: per-row input norms (one wave per row) ----------------
__global__ void __launch_bounds__(256) nx_kernel(const float* __restrict__ inputs,
                                                 float* __restrict__ nx) {
    const int wave = threadIdx.x >> 5, lane = threadIdx.x & 31;
    const int row = blockIdx.x * 8 + wave;
    const float4* p = (const float4*)(inputs + (size_t)row * DIM + lane * 8);
    const float4 v0 = p[0], v1 = p[1];
    float sq = v0.x * v0.x + v0.y * v0.y + v0.z * v0.z + v0.w * v0.w +
               v1.x * v1.x + v1.y * v1.y + v1.z * v1.z + v1.w * v1.w;
    for (int off = 16; off > 0; off >>= 1) sq += __shfl_xor(sq, off, 32);
    if (lane == 0) nx[row] = sqrtf(sq);
}

// ---------------- K4: split-bf16 WMMA GEMM + cosine scaling + fused online softmax CE ----------------
// grid = 512 (128 rows each), block = 256 (8 waves, 16 rows per wave)
#define BSTRIDE 280   // bf16 elements per LDS row: 560 B, 16B-aligned, near-conflict-free b128 reads
__global__ void __launch_bounds__(256) wmma_ce_kernel(const float* __restrict__ inputs,
                                                      const int* __restrict__ targets,
                                                      const __bf16* __restrict__ phi,
                                                      const __bf16* __restrict__ plo,
                                                      const float* __restrict__ npn,
                                                      const float* __restrict__ nx,
                                                      float* __restrict__ out) {
    __shared__ __bf16 ldsHL[2 * 16 * BSTRIDE];
    __bf16* ldsH = ldsHL;
    __bf16* ldsL = ldsHL + 16 * BSTRIDE;

    const int tid  = threadIdx.x;
    const int wave = tid >> 5;
    const int lane = tid & 31;
    const int hl   = lane >> 4;       // half-wave: selects K sub-offsets in the fragments
    const int l16  = lane & 15;       // M (A) / N (B) index within a 16-wide tile
    const int rowBase = blockIdx.x * 128 + wave * 16;

    // A panel, split into bf16 hi/lo fragments, kept in registers for all 32 column tiles.
    // 16x32 bf16 A layout: lanes<16 -> K {0..7,16..23}; lanes>=16 -> K {8..15,24..31}.
    v16bf ahi[8], alo[8];
    const float* arow = inputs + (size_t)(rowBase + l16) * DIM;
#pragma unroll
    for (int kk = 0; kk < 8; ++kk) {
        const float* p = arow + kk * 32 + hl * 8;
        const float4 x0 = *(const float4*)(p);
        const float4 x1 = *(const float4*)(p + 4);
        const float4 x2 = *(const float4*)(p + 16);
        const float4 x3 = *(const float4*)(p + 20);
        const float xs[16] = {x0.x, x0.y, x0.z, x0.w, x1.x, x1.y, x1.z, x1.w,
                              x2.x, x2.y, x2.z, x2.w, x3.x, x3.y, x3.z, x3.w};
#pragma unroll
        for (int j = 0; j < 16; ++j) {
            const unsigned short hb = bfbits_trunc(xs[j]);
            ahi[kk][j] = u2bf(hb);
            alo[kk][j] = f2bf_rne(xs[j] - __uint_as_float((unsigned)hb << 16));
        }
    }

    float nxr[8], m[8], s[8], tv[8];
    int tg[8];
#pragma unroll
    for (int r = 0; r < 8; ++r) {
        const int row = rowBase + hl * 8 + r;   // C layout: VGPR r holds row hl*8+r
        nxr[r] = nx[row];
        tg[r]  = targets[row];
        m[r]   = -1e30f;
        s[r]   = 0.0f;
        tv[r]  = 0.0f;
    }

    for (int ct = 0; ct < 32; ++ct) {
        const int c0 = ct * 16;
        __syncthreads();
        {   // stage 16 prototype rows (hi + lo bf16 planes) into padded LDS: 32 B per thread per plane
            const int cr  = tid >> 4;
            const int seg = (tid & 15) * 16;          // 16 bf16 = 32 B
            const uint4* sH = (const uint4*)(phi + (size_t)(c0 + cr) * DIM + seg);
            const uint4* sL = (const uint4*)(plo + (size_t)(c0 + cr) * DIM + seg);
            uint4* dH = (uint4*)(ldsH + cr * BSTRIDE + seg);
            uint4* dL = (uint4*)(ldsL + cr * BSTRIDE + seg);
            dH[0] = sH[0]; dH[1] = sH[1];
            dL[0] = sL[0]; dL[1] = sL[1];
        }
        __syncthreads();

        const float npc = npn[c0 + l16];
        v8f accv = {0.f, 0.f, 0.f, 0.f, 0.f, 0.f, 0.f, 0.f};
        // 32x16 bf16 B layout: lanes<16 -> K {0..7,16..23}; lanes>=16 -> K {8..15,24..31}.
        const __bf16* bHrow = ldsH + l16 * BSTRIDE + hl * 8;
        const __bf16* bLrow = ldsL + l16 * BSTRIDE + hl * 8;
#pragma unroll
        for (int kk = 0; kk < 8; ++kk) {
            const v8bf h0 = *(const v8bf*)(bHrow + kk * 32);
            const v8bf h1 = *(const v8bf*)(bHrow + kk * 32 + 16);
            const v8bf l0 = *(const v8bf*)(bLrow + kk * 32);
            const v8bf l1 = *(const v8bf*)(bLrow + kk * 32 + 16);
            const v16bf bh = __builtin_shufflevector(h0, h1, 0,1,2,3,4,5,6,7,8,9,10,11,12,13,14,15);
            const v16bf bl = __builtin_shufflevector(l0, l1, 0,1,2,3,4,5,6,7,8,9,10,11,12,13,14,15);
            accv = __builtin_amdgcn_wmma_f32_16x16x32_bf16(false, ahi[kk], false, bh, (short)0, accv, false, false);
            accv = __builtin_amdgcn_wmma_f32_16x16x32_bf16(false, ahi[kk], false, bl, (short)0, accv, false, false);
            accv = __builtin_amdgcn_wmma_f32_16x16x32_bf16(false, alo[kk], false, bh, (short)0, accv, false, false);
        }

        const int colc = c0 + l16;
#pragma unroll
        for (int r = 0; r < 8; ++r) {
            const float denom = fmaxf(nxr[r] * npc, 1e-8f);
            const float sim = accv[r] / denom;
            const float mn = fmaxf(m[r], sim);
            s[r] = s[r] * __expf(m[r] - mn) + __expf(sim - mn);
            m[r] = mn;
            tv[r] += (tg[r] == colc) ? sim : 0.0f;
        }
    }

    // combine (max, sumexp, target-sim) across the 16 lanes that share each row
#pragma unroll
    for (int off = 1; off < 16; off <<= 1) {
#pragma unroll
        for (int r = 0; r < 8; ++r) {
            const float mo = __shfl_xor(m[r], off, 32);
            const float so = __shfl_xor(s[r], off, 32);
            const float to = __shfl_xor(tv[r], off, 32);
            const float mn = fmaxf(m[r], mo);
            s[r] = s[r] * __expf(m[r] - mn) + so * __expf(mo - mn);
            m[r] = mn;
            tv[r] += to;
        }
    }
    if (l16 == 0) {   // lanes 0 and 16 carry rows 0..7 and 8..15 respectively
        float partial = 0.0f;
#pragma unroll
        for (int r = 0; r < 8; ++r)
            partial += m[r] + __logf(s[r]) - tv[r];   // logsumexp - sim[target]
        atomicAdd(out, partial * (1.0f / (float)N_ROWS));
    }
}

extern "C" void kernel_launch(void* const* d_in, const int* in_sizes, int n_in,
                              void* d_out, int out_size, void* d_ws, size_t ws_size,
                              hipStream_t stream) {
    const float* inputs  = (const float*)d_in[0];
    const int*   targets = (const int*)d_in[1];
    float* out = (float*)d_out;
    float* ws  = (float*)d_ws;

    // workspace layout (float-sized offsets)
    float*  sums   = ws;                         // 131072 f32
    float*  counts = ws + 131072;                // 512 f32
    float*  npn    = ws + 131584;                // 512 f32
    float*  nx     = ws + 132096;                // 65536 f32
    __bf16* phi    = (__bf16*)(ws + 197632);     // 131072 bf16
    __bf16* plo    = (__bf16*)(ws + 263168);     // 131072 bf16
    (void)in_sizes; (void)n_in; (void)out_size; (void)ws_size;

    zero_kernel<<<(131584 + 255) / 256, 256, 0, stream>>>(sums, 131584, out);

    dim3 sg(16, 16);
    scatter_kernel<<<sg, 256, 0, stream>>>(inputs, targets, sums);
    count_kernel<<<64, 256, 0, stream>>>(targets, counts);
    proto_kernel<<<NCLS, 256, 0, stream>>>(sums, counts, phi, plo, npn);
    nx_kernel<<<N_ROWS / 8, 256, 0, stream>>>(inputs, nx);
    wmma_ce_kernel<<<N_ROWS / 128, 256, 0, stream>>>(inputs, targets, phi, plo, npn, nx, out);
}
